// ConceptQuerySlotAttention_39393440039465
// MI455X (gfx1250) — compile-verified, matrix operand order checked by reading, not run on hardware
//
#include <hip/hip_runtime.h>

// ---------------------------------------------------------------------------
// Slot attention (B=16, N=4096, S=16, D=512, 3 iters) for gfx1250.
// All GEMMs: bf16 WMMA (v_wmma_f32_16x16x32_bf16), fp32 accumulation.
// Wave tile: (16*MT) x 128; MT=2 for the big GEMMs so each B fragment feeds
// two WMMAs (halves B-operand traffic per FLOP), MT=1 for the M=16 batched
// attention GEMMs.  Fragment loads and the transposed C^T store are b128; row-major C stores are per element.
// V is stored transposed so every GEMM uses the fast row-major B loader.
// ---------------------------------------------------------------------------

typedef __bf16 v16bf __attribute__((ext_vector_type(16)));
typedef float v8f __attribute__((ext_vector_type(8)));
typedef unsigned short ushort8 __attribute__((ext_vector_type(8)));

__device__ __forceinline__ unsigned short f2bf(float f) {
    unsigned int u = __float_as_uint(f);
    unsigned int r = u + 0x7FFFu + ((u >> 16) & 1u);   // round-to-nearest-even
    return (unsigned short)(r >> 16);
}

// 16x32 bf16 fragment from a row-major matrix (leading dim lda, elements).
// ISA 7.12.2 layout: lanes 0-15 -> rows, K 0..7 + 16..23; lanes 16-31 ->
// same rows, K 8..15 + 24..31.  Two 16-byte vector loads per lane.
__device__ __forceinline__ v16bf load_frag_rowmajor(const unsigned short* __restrict__ A,
                                                    int lda, int row0, int k0, int lane) {
    const unsigned short* p =
        A + (size_t)(row0 + (lane & 15)) * lda + (k0 + ((lane >> 4) << 3));
    union { ushort8 h[2]; v16bf v; } f;
    f.h[0] = *(const ushort8*)(p);
    f.h[1] = *(const ushort8*)(p + 16);
    return f.v;
}

// C = A[M,K] * W^T with W[N,K] row-major.  One wave per (16*MT) x 128 C tile.
// Cf      : optional fp32 row-major output (ldc)
// Cb      : optional bf16 output; storeT==0 -> row-major (ldc),
//           storeT==1 -> transposed C^T[N,M] with leading dim ldct
//           (per-lane packed ushort8 -> global_store_b128).
template <int MT>
__global__ __launch_bounds__(32) void gemm_bf16_wmma(
    const unsigned short* __restrict__ A, const unsigned short* __restrict__ W,
    float* __restrict__ Cf, unsigned short* __restrict__ Cb,
    int K, int lda, int ldb, int ldc, int ldct, int storeT,
    long long sA, long long sB, long long sC) {
    const int lane = threadIdx.x;
    const int bz = blockIdx.z;
    A += (long long)bz * sA;
    W += (long long)bz * sB;
    const long long coff = (long long)bz * sC;
    const int row0 = (blockIdx.y * MT) << 4;
    const int col0 = blockIdx.x << 7;   // 128 columns per wave

    v8f acc[MT][8];
#pragma unroll
    for (int m = 0; m < MT; ++m)
#pragma unroll
        for (int t = 0; t < 8; ++t) acc[m][t] = v8f{};

    for (int k0 = 0; k0 < K; k0 += 32) {
        v16bf a[MT];
#pragma unroll
        for (int m = 0; m < MT; ++m)
            a[m] = load_frag_rowmajor(A, lda, row0 + 16 * m, k0, lane);
#pragma unroll
        for (int t = 0; t < 8; ++t) {
            const v16bf b = load_frag_rowmajor(W, ldb, col0 + 16 * t, k0, lane);
#pragma unroll
            for (int m = 0; m < MT; ++m)
                acc[m][t] = __builtin_amdgcn_wmma_f32_16x16x32_bf16(
                    false, a[m], false, b, (short)0, acc[m][t], false, false);
        }
    }

    const int nl = lane & 15;
#pragma unroll
    for (int m = 0; m < MT; ++m) {
        const int mb = row0 + 16 * m + ((lane >> 4) << 3);  // first of 8 rows
        if (storeT) {
            // C^T[col][row]: 8 row-values are contiguous per lane -> b128 store
#pragma unroll
            for (int t = 0; t < 8; ++t) {
                const int col = col0 + 16 * t + nl;
                ushort8 pk;
#pragma unroll
                for (int j = 0; j < 8; ++j) pk[j] = f2bf(acc[m][t][j]);
                *(ushort8*)(Cb + coff + (size_t)col * ldct + mb) = pk;
            }
        } else {
#pragma unroll
            for (int j = 0; j < 8; ++j) {
                const long long rbase = coff + (long long)(mb + j) * ldc + col0 + nl;
#pragma unroll
                for (int t = 0; t < 8; ++t) {
                    const long long idx = rbase + (t << 4);
                    const float v = acc[m][t][j];
                    if (Cf) Cf[idx] = v;
                    if (Cb) Cb[idx] = f2bf(v);
                }
            }
        }
    }
}

// ------------------------- LayerNorm (rows of length D) --------------------
__device__ __forceinline__ float wave_reduce_add(float v) {
#pragma unroll
    for (int off = 16; off > 0; off >>= 1) v += __shfl_down(v, off, 32);
    return v;
}

__global__ __launch_bounds__(256) void layernorm_rows(
    const float* __restrict__ x, const float* __restrict__ w,
    const float* __restrict__ bias, unsigned short* __restrict__ y, int D) {
    const int row = blockIdx.x;
    const float* xr = x + (long long)row * D;
    float sum = 0.f, sumsq = 0.f;
    for (int i = threadIdx.x; i < D; i += blockDim.x) {
        const float v = xr[i];
        sum += v;
        sumsq += v * v;
    }
    __shared__ float lsum[8], lsq[8];
    __shared__ float s_mu, s_rstd;
    const int wid = threadIdx.x >> 5, lid = threadIdx.x & 31;
    const float ws = wave_reduce_add(sum);
    const float wq = wave_reduce_add(sumsq);
    if (lid == 0) { lsum[wid] = ws; lsq[wid] = wq; }
    __syncthreads();
    if (threadIdx.x == 0) {
        float S = 0.f, Q = 0.f;
        for (int i = 0; i < 8; ++i) { S += lsum[i]; Q += lsq[i]; }
        const float m = S / (float)D;
        const float var = Q / (float)D - m * m;
        s_mu = m;
        s_rstd = rsqrtf(var + 1e-5f);
    }
    __syncthreads();
    const float mu = s_mu, rstd = s_rstd;
    for (int i = threadIdx.x; i < D; i += blockDim.x)
        y[(long long)row * D + i] = f2bf((xr[i] - mu) * rstd * w[i] + bias[i]);
}

// ------------------------- fp32 -> bf16 conversion -------------------------
__global__ __launch_bounds__(256) void cvt_f32_bf16(const float* __restrict__ x,
                                                    unsigned short* __restrict__ y,
                                                    long long n) {
    const long long i = (long long)blockIdx.x * blockDim.x + threadIdx.x;
    if (i < n) y[i] = f2bf(x[i]);
}

// ---------------- softmax over slot axis + row-sum accumulation ------------
__global__ __launch_bounds__(256) void softmax_over_slots(
    float* __restrict__ attn, float* __restrict__ rowsum, int N, float scale) {
    const int b = blockIdx.y;
    const int n = blockIdx.x * blockDim.x + threadIdx.x;
    __shared__ float part[16];
    if (threadIdx.x < 16) part[threadIdx.x] = 0.f;
    __syncthreads();
    const long long base = (long long)b * 16 * N + n;
    float v[16];
    float mx = -1e30f;
#pragma unroll
    for (int s = 0; s < 16; ++s) {
        v[s] = attn[base + (long long)s * N] * scale;
        mx = fmaxf(mx, v[s]);
    }
    float sum = 0.f;
#pragma unroll
    for (int s = 0; s < 16; ++s) {
        v[s] = __expf(v[s] - mx);
        sum += v[s];
    }
    const float inv = 1.f / sum;
#pragma unroll
    for (int s = 0; s < 16; ++s) {
        const float a = v[s] * inv;
        attn[base + (long long)s * N] = a;
        atomicAdd(&part[s], a);
    }
    __syncthreads();
    if (threadIdx.x < 16) atomicAdd(&rowsum[b * 16 + threadIdx.x], part[threadIdx.x]);
}

// attn_wm = attn / (rowsum + eps), stored bf16 for the updates GEMM.
__global__ __launch_bounds__(256) void weighted_mean_prep(
    const float* __restrict__ attn, const float* __restrict__ rowsum,
    unsigned short* __restrict__ wm, int N, long long total) {
    const long long i = (long long)blockIdx.x * blockDim.x + threadIdx.x;
    if (i < total) {
        const long long bs = i / N;
        wm[i] = f2bf(attn[i] / (rowsum[bs] + 1e-8f));
    }
}

// -------------------------------- GRU gates --------------------------------
__global__ __launch_bounds__(256) void gru_combine(
    const float* __restrict__ gi, const float* __restrict__ gh,
    const float* __restrict__ b_ih, const float* __restrict__ b_hh,
    float* __restrict__ h, int D, long long total) {
    const long long i = (long long)blockIdx.x * blockDim.x + threadIdx.x;
    if (i >= total) return;
    const int d = (int)(i % D);
    const long long row = i / D;
    const float* gir = gi + row * 3 * D;
    const float* ghr = gh + row * 3 * D;
    const float ir = gir[d] + b_ih[d];
    const float iz = gir[D + d] + b_ih[D + d];
    const float in_ = gir[2 * D + d] + b_ih[2 * D + d];
    const float hr = ghr[d] + b_hh[d];
    const float hz = ghr[D + d] + b_hh[D + d];
    const float hn = ghr[2 * D + d] + b_hh[2 * D + d];
    const float r = 1.f / (1.f + __expf(-(ir + hr)));
    const float z = 1.f / (1.f + __expf(-(iz + hz)));
    const float nn = tanhf(in_ + r * hn);
    h[i] = (1.f - z) * nn + z * h[i];
}

// ---------------------------------------------------------------------------
static void launch_gemm(const unsigned short* A, const unsigned short* W,
                        float* Cf, unsigned short* Cb,
                        int M, int N, int K, int lda, int ldb, int ldc,
                        int ldct, int storeT,
                        long long sA, long long sB, long long sC,
                        int batches, int MT, hipStream_t st) {
    if (MT == 2) {
        dim3 grid(N / 128, M / 32, batches);
        gemm_bf16_wmma<2><<<grid, 32, 0, st>>>(A, W, Cf, Cb, K, lda, ldb, ldc,
                                               ldct, storeT, sA, sB, sC);
    } else {
        dim3 grid(N / 128, M / 16, batches);
        gemm_bf16_wmma<1><<<grid, 32, 0, st>>>(A, W, Cf, Cb, K, lda, ldb, ldc,
                                               ldct, storeT, sA, sB, sC);
    }
}

extern "C" void kernel_launch(void* const* d_in, const int* in_sizes, int n_in,
                              void* d_out, int out_size, void* d_ws, size_t ws_size,
                              hipStream_t stream) {
    (void)in_sizes; (void)n_in; (void)out_size; (void)ws_size;

    constexpr int B = 16, N = 4096, S = 16, D = 512;
    constexpr int BN = B * N;     // 65536 feature rows
    constexpr int BS = B * S;     // 256 slot rows
    const float scale = 0.044194173824159216f;  // D^-0.5

    const float* features   = (const float*)d_in[0];
    const float* slots_init = (const float*)d_in[1];
    const float* ln_f_w = (const float*)d_in[2];
    const float* ln_f_b = (const float*)d_in[3];
    const float* ln_s_w = (const float*)d_in[4];
    const float* ln_s_b = (const float*)d_in[5];
    const float* Wk  = (const float*)d_in[6];
    const float* Wv  = (const float*)d_in[7];
    const float* Wq  = (const float*)d_in[8];
    const float* Wih = (const float*)d_in[9];
    const float* Whh = (const float*)d_in[10];
    const float* b_ih = (const float*)d_in[11];
    const float* b_hh = (const float*)d_in[12];

    // ---- workspace carving (256B aligned) ----
    size_t off = 0;
    auto carve = [&](size_t bytes) -> void* {
        off = (off + 255) & ~(size_t)255;
        void* p = (char*)d_ws + off;
        off += bytes;
        return p;
    };
    unsigned short* feats_bf = (unsigned short*)carve((size_t)BN * D * 2);
    unsigned short* k_bf     = (unsigned short*)carve((size_t)BN * D * 2);  // [B][N][D]
    unsigned short* vt_bf    = (unsigned short*)carve((size_t)BN * D * 2);  // [B][D][N]
    unsigned short* Wk_bf    = (unsigned short*)carve((size_t)D * D * 2);
    unsigned short* Wv_bf    = (unsigned short*)carve((size_t)D * D * 2);
    unsigned short* Wq_bf    = (unsigned short*)carve((size_t)D * D * 2);
    unsigned short* Wih_bf   = (unsigned short*)carve((size_t)3 * D * D * 2);
    unsigned short* Whh_bf   = (unsigned short*)carve((size_t)3 * D * D * 2);
    float*          slots    = (float*)carve((size_t)BS * D * 4);
    unsigned short* slotsn_bf= (unsigned short*)carve((size_t)BS * D * 2);
    unsigned short* slotsp_bf= (unsigned short*)carve((size_t)BS * D * 2);
    unsigned short* q_bf     = (unsigned short*)carve((size_t)BS * D * 2);
    float*          attn     = (float*)carve((size_t)B * S * N * 4);
    float*          rowsum   = (float*)carve((size_t)BS * 4);
    unsigned short* wm_bf    = (unsigned short*)carve((size_t)B * S * N * 2);
    float*          updates  = (float*)carve((size_t)BS * D * 4);
    unsigned short* upd_bf   = (unsigned short*)carve((size_t)BS * D * 2);
    float*          gi       = (float*)carve((size_t)BS * 3 * D * 4);
    float*          gh       = (float*)carve((size_t)BS * 3 * D * 4);

    // ---- weights -> bf16 ----
    cvt_f32_bf16<<<(D * D + 255) / 256, 256, 0, stream>>>(Wk, Wk_bf, (long long)D * D);
    cvt_f32_bf16<<<(D * D + 255) / 256, 256, 0, stream>>>(Wv, Wv_bf, (long long)D * D);
    cvt_f32_bf16<<<(D * D + 255) / 256, 256, 0, stream>>>(Wq, Wq_bf, (long long)D * D);
    cvt_f32_bf16<<<(3 * D * D + 255) / 256, 256, 0, stream>>>(Wih, Wih_bf, (long long)3 * D * D);
    cvt_f32_bf16<<<(3 * D * D + 255) / 256, 256, 0, stream>>>(Whh, Whh_bf, (long long)3 * D * D);

    // ---- feats = LN(features) ; k = feats Wk^T ; v^T (transposed store) ----
    layernorm_rows<<<BN, 256, 0, stream>>>(features, ln_f_w, ln_f_b, feats_bf, D);
    // k: single GEMM over all BN rows, row-major bf16 out (MT=2)
    launch_gemm(feats_bf, Wk_bf, nullptr, k_bf, BN, D, D, D, D, D, 0, 0,
                0, 0, 0, 1, 2, stream);
    // v: batched per b so the transposed tile lands in v_t[b] = [D][N] (MT=2)
    launch_gemm(feats_bf, Wv_bf, nullptr, vt_bf, N, D, D, D, D, D, N, 1,
                (long long)N * D, 0, (long long)D * N, B, 2, stream);

    // ---- slots = slots_init ----
    hipMemcpyAsync(slots, slots_init, (size_t)BS * D * 4, hipMemcpyDeviceToDevice, stream);

    for (int it = 0; it < 3; ++it) {
        layernorm_rows<<<BS, 256, 0, stream>>>(slots, ln_s_w, ln_s_b, slotsn_bf, D);
        cvt_f32_bf16<<<(BS * D + 255) / 256, 256, 0, stream>>>(slots, slotsp_bf, (long long)BS * D);

        // q = slots_n Wq^T  [256, 512] (MT=2)
        launch_gemm(slotsn_bf, Wq_bf, nullptr, q_bf, BS, D, D, D, D, D, 0, 0,
                    0, 0, 0, 1, 2, stream);

        // logits[b] = q[b] k[b]^T  -> [16, 4096] (MT=1, M=16)
        launch_gemm(q_bf, k_bf, attn, nullptr, S, N, D, D, D, N, 0, 0,
                    (long long)S * D, (long long)N * D, (long long)S * N, B, 1, stream);

        hipMemsetAsync(rowsum, 0, (size_t)BS * 4, stream);
        softmax_over_slots<<<dim3(N / 256, B, 1), 256, 0, stream>>>(attn, rowsum, N, scale);

        weighted_mean_prep<<<(B * S * N + 255) / 256, 256, 0, stream>>>(
            attn, rowsum, wm_bf, N, (long long)B * S * N);

        // updates[b] = attn_wm[b] (v_t[b])^T -> [16, 512] (MT=1, M=16)
        launch_gemm(wm_bf, vt_bf, updates, upd_bf, S, D, N, N, N, D, 0, 0,
                    (long long)S * N, (long long)D * N, (long long)S * D, B, 1, stream);

        // GRU: gi = updates W_ih^T, gh = slots_prev W_hh^T -> [256, 1536] (MT=2)
        launch_gemm(upd_bf, Wih_bf, gi, nullptr, BS, 3 * D, D, D, D, 3 * D, 0, 0,
                    0, 0, 0, 1, 2, stream);
        launch_gemm(slotsp_bf, Whh_bf, gh, nullptr, BS, 3 * D, D, D, D, 3 * D, 0, 0,
                    0, 0, 0, 1, 2, stream);
        gru_combine<<<(BS * D + 255) / 256, 256, 0, stream>>>(
            gi, gh, b_ih, b_hh, slots, D, (long long)BS * D);
    }

    // outputs: slots [B,S,D] then attn [B,S,N] (fp32, concatenated)
    float* out = (float*)d_out;
    hipMemcpyAsync(out, slots, (size_t)BS * D * 4, hipMemcpyDeviceToDevice, stream);
    hipMemcpyAsync(out + (size_t)BS * D, attn, (size_t)B * S * N * 4,
                   hipMemcpyDeviceToDevice, stream);
}
